// TFReformerLM_22771916603626
// MI455X (gfx1250) — matrix operand order repaired and checked by reading, not compiled, not run on hardware
//
#include <hip/hip_runtime.h>

// ---------------- model constants ----------------
#define SLEN   4096
#define EMBD   512
#define NHEAD  8
#define DHEAD  64
#define NHASH  4
#define BS     64          // bucket size
#define NBUCK  64          // SLEN / BS
#define NCH    256         // NHASH * NBUCK chunks per head
#define TOT    16384       // NHASH * SLEN
#define FFND   2048
#define VOCAB  32000

typedef __attribute__((ext_vector_type(16))) __bf16 v16bf;
typedef __attribute__((ext_vector_type(8)))  __bf16 v8bf;
typedef __attribute__((ext_vector_type(8)))  float  v8f;

union Frag16 { v16bf v; v8bf h[2]; };

__device__ __forceinline__ v8f wmma_bf16(v16bf a, v16bf b, v8f c) {
    return __builtin_amdgcn_wmma_f32_16x16x32_bf16(false, a, false, b, (short)0, c, false, false);
}

// ---------------- embedding ----------------
__global__ __launch_bounds__(256) void embed_kernel(const int* __restrict__ tok,
                                                    const float* __restrict__ te,
                                                    const float* __restrict__ pe,
                                                    float* __restrict__ x1,
                                                    float* __restrict__ x2) {
    int i = blockIdx.x * 256 + threadIdx.x;          // SLEN*EMBD threads
    int s = i >> 9;
    int e = i & (EMBD - 1);
    float v = te[(size_t)tok[s] * EMBD + e] + pe[i];
    x1[i] = v; x2[i] = v;
}

// ---------------- layernorm (f32 in -> bf16 out) ----------------
__global__ __launch_bounds__(256) void layernorm_bf16(const float* __restrict__ x,
                                                      const float* __restrict__ g,
                                                      const float* __restrict__ b,
                                                      __bf16* __restrict__ out) {
    int row = blockIdx.x;
    const float* xr = x + (size_t)row * EMBD;
    int tid = threadIdx.x;
    float x0 = xr[tid], x1v = xr[tid + 256];
    float s = x0 + x1v, s2 = x0 * x0 + x1v * x1v;
    for (int o = 1; o < 32; o <<= 1) { s += __shfl_xor(s, o, 32); s2 += __shfl_xor(s2, o, 32); }
    __shared__ float rs[8], rs2[8];
    __shared__ float smean, srstd;
    int wave = tid >> 5, lane = tid & 31;
    if (lane == 0) { rs[wave] = s; rs2[wave] = s2; }
    __syncthreads();
    if (tid == 0) {
        float ts = 0.f, ts2 = 0.f;
        for (int i = 0; i < 8; i++) { ts += rs[i]; ts2 += rs2[i]; }
        float m = ts * (1.f / EMBD);
        float v = ts2 * (1.f / EMBD) - m * m;
        smean = m; srstd = rsqrtf(v + 1e-3f);
    }
    __syncthreads();
    float m = smean, r = srstd;
    out[(size_t)row * EMBD + tid]       = (__bf16)((x0  - m) * r * g[tid]       + b[tid]);
    out[(size_t)row * EMBD + tid + 256] = (__bf16)((x1v - m) * r * g[tid + 256] + b[tid + 256]);
}

// ---------------- weight pack: f32 [K,N] -> bf16 B-fragment layout ----------------
// packed[((nt*ktiles + kt)*32 + lane)*16 + i] = W[kt*32 + (lane>>4)*16 + i][nt*16 + (lane&15)]
__global__ __launch_bounds__(256) void pack_kernel(const float* __restrict__ W,
                                                   __bf16* __restrict__ P,
                                                   int K, int N) {
    int tid = blockIdx.x * 256 + threadIdx.x;
    if (tid >= K * N) return;
    int i    = tid & 15;
    int lane = (tid >> 4) & 31;
    int tile = tid >> 9;
    int ktiles = K >> 5;
    int kt = tile % ktiles;
    int nt = tile / ktiles;
    int k = kt * 32 + (lane >> 4) * 16 + i;
    int n = nt * 16 + (lane & 15);
    P[tid] = (__bf16)W[(size_t)k * N + n];
}

// ---------------- generic WMMA GEMM: out = act(A@B + bias) [+ res] ----------------
// A: bf16 row-major [M,K]; Bp: packed bf16; block tile 64(M) x 128(N)
__global__ __launch_bounds__(256) void wmma_gemm(const __bf16* __restrict__ A,
                                                 const __bf16* __restrict__ Bp,
                                                 const float* __restrict__ bias,
                                                 const float* __restrict__ res,
                                                 float* __restrict__ outF,
                                                 __bf16* __restrict__ outB,
                                                 int N, int K, int doGelu) {
    int wave = threadIdx.x >> 5, lane = threadIdx.x & 31;
    int wm = wave & 3, wn = wave >> 2;
    int mBase  = blockIdx.y * 64 + wm * 16;
    int ntile0 = blockIdx.x * 8 + wn * 4;
    int ktiles = K >> 5;
    int half = lane >> 4;
    v8f zero = {0.f,0.f,0.f,0.f,0.f,0.f,0.f,0.f};
    v8f acc[4] = {zero, zero, zero, zero};
    const __bf16* arow = A + (size_t)(mBase + (lane & 15)) * K;
    for (int kt = 0; kt < ktiles; kt++) {
        Frag16 a;
        a.v = *(const v16bf*)(arow + kt * 32 + half * 16);
#pragma unroll
        for (int t = 0; t < 4; t++) {
            Frag16 b;
            b.v = *(const v16bf*)(Bp + ((size_t)((ntile0 + t) * ktiles + kt) * 32 + lane) * 16);
            acc[t] = wmma_bf16(a.v, b.v, acc[t]);
        }
    }
#pragma unroll
    for (int t = 0; t < 4; t++) {
        int col = (ntile0 + t) * 16 + (lane & 15);
        float bv = bias ? bias[col] : 0.f;
#pragma unroll
        for (int r = 0; r < 8; r++) {
            int m = mBase + half * 8 + r;
            float v = acc[t][r] + bv;
            if (doGelu) {
                float x = v;
                v = 0.5f * x * (1.f + tanhf(0.7978845608f * (x + 0.044715f * x * x * x)));
            }
            size_t o = (size_t)m * N + col;
            if (res)  v += res[o];
            if (outF) outF[o] = v;
            if (outB) outB[o] = (__bf16)v;
        }
    }
}

// ---------------- LSH bucketing ----------------
__global__ __launch_bounds__(256) void bucket_kernel(const float* __restrict__ QK,
                                                     const float* __restrict__ rot,
                                                     int* __restrict__ buckets) {
    int i = blockIdx.x * 256 + threadIdx.x;     // NHEAD*NHASH*SLEN threads
    int s  = i & (SLEN - 1);
    int h  = (i >> 12) & (NHASH - 1);
    int bh = i >> 14;
    float acc[32];
#pragma unroll
    for (int r = 0; r < 32; r++) acc[r] = 0.f;
    const float* q = QK + (size_t)s * EMBD + bh * DHEAD;
    for (int d = 0; d < DHEAD; d++) {
        float qv = q[d];
        const float* rp = rot + (size_t)(d * NHASH + h) * 32;
#pragma unroll
        for (int r = 0; r < 32; r++) acc[r] += qv * rp[r];
    }
    float best = acc[0]; int bi = 0;
    for (int r = 1; r < 32; r++) if (acc[r] > best) { best = acc[r]; bi = r; }
    for (int r = 0; r < 32; r++) if (-acc[r] > best) { best = -acc[r]; bi = 32 + r; }
    buckets[(size_t)bh * TOT + h * SLEN + s] = bi + h * NBUCK;
}

// ---------------- stable counting sort per head (keys: bucket then pos) ----------------
__global__ __launch_bounds__(256) void sort_kernel(const int* __restrict__ buckets,
                                                   int* __restrict__ sorted,
                                                   int* __restrict__ undo) {
    int bh = blockIdx.x;
    const int* bk = buckets + (size_t)bh * TOT;
    __shared__ int hist[256];
    __shared__ int offs[256];
    int t = threadIdx.x;
    hist[t] = 0;
    __syncthreads();
    for (int i = t; i < TOT; i += 256) atomicAdd(&hist[bk[i]], 1);
    __syncthreads();
    if (t == 0) { int a = 0; for (int b = 0; b < 256; b++) { offs[b] = a; a += hist[b]; } }
    __syncthreads();
    int b = t;
    int off = offs[b];
    for (int i = 0; i < TOT; i++) {
        if (bk[i] == b) {
            sorted[(size_t)bh * TOT + off] = i;
            undo[(size_t)bh * TOT + i] = off;
            off++;
        }
    }
}

// ---------------- chunked LSH attention (one block per (chunk, head)) ----------------
__global__ __launch_bounds__(256) void attn_kernel(const float* __restrict__ QK,
                                                   const float* __restrict__ V,
                                                   const int* __restrict__ sorted,
                                                   float* __restrict__ so,
                                                   float* __restrict__ slog) {
    int c  = blockIdx.x;
    int bh = blockIdx.y;
    __shared__ __align__(64) __bf16 s_bq[64][64];
    __shared__ __align__(64) __bf16 s_bk[128][64];
    __shared__ __align__(64) __bf16 s_bvT[64][128];
    __shared__ __align__(64) __bf16 s_probs[64][128];
    __shared__ int s_bqt[64];
    __shared__ int s_bkvt[128];

    int tid = threadIdx.x;
    // ---- gather sorted rows into LDS ----
    {
        int j   = tid & 127;
        int isV = tid >> 7;
        int cc  = (j < 64) ? c : ((c + NCH - 1) & (NCH - 1));
        int pos = cc * BS + (j & 63);
        int t   = sorted[(size_t)bh * TOT + pos];
        int sp  = t & (SLEN - 1);
        if (!isV) {
            const float* row = QK + (size_t)sp * EMBD + bh * DHEAD;
            float ss = 0.f;
            for (int e = 0; e < DHEAD; e++) { float x = row[e]; ss += x * x; }
            float inv = rsqrtf(ss + 1e-12f);
            for (int e = 0; e < DHEAD; e++) {
                float x = row[e];
                s_bk[j][e] = (__bf16)(x * inv);
                if (j < 64) s_bq[j][e] = (__bf16)x;
            }
            s_bkvt[j] = sp;
            if (j < 64) s_bqt[j] = sp;
        } else {
            const float* row = V + (size_t)sp * EMBD + bh * DHEAD;
            for (int e = 0; e < DHEAD; e++) s_bvT[e][j] = (__bf16)row[e];
        }
    }
    __syncthreads();

    int wave = tid >> 5, lane = tid & 31;
    int half = lane >> 4;
    v8f zero = {0.f,0.f,0.f,0.f,0.f,0.f,0.f,0.f};

    // ---- dots = bq @ bk^T (64x128, K=64), mask, lse, probs ----
    if (wave < 4) {
        v8f acc[8] = {zero,zero,zero,zero,zero,zero,zero,zero};
        int mrow = wave * 16 + (lane & 15);
        for (int kb = 0; kb < 64; kb += 32) {
            Frag16 a;
            a.v = *(const v16bf*)&s_bq[mrow][kb + half * 16];
#pragma unroll
            for (int jt = 0; jt < 8; jt++) {
                Frag16 b;
                b.v = *(const v16bf*)&s_bk[jt * 16 + (lane & 15)][kb + half * 16];
                acc[jt] = wmma_bf16(a.v, b.v, acc[jt]);
            }
        }
        // scale + self-mask
#pragma unroll
        for (int jt = 0; jt < 8; jt++)
#pragma unroll
            for (int r = 0; r < 8; r++) {
                int ml = wave * 16 + half * 8 + r;
                int jc = jt * 16 + (lane & 15);
                float v = acc[jt][r] * 0.125f;
                if (s_bqt[ml] == s_bkvt[jc]) v = -5e4f;
                acc[jt][r] = v;
            }
        // per-row logsumexp over 128 cols (8 tiles x 16 lanes in same half)
#pragma unroll
        for (int r = 0; r < 8; r++) {
            float mx = -3.4e38f;
#pragma unroll
            for (int jt = 0; jt < 8; jt++) mx = fmaxf(mx, acc[jt][r]);
            for (int o = 1; o < 16; o <<= 1) mx = fmaxf(mx, __shfl_xor(mx, o, 32));
            float sm = 0.f;
#pragma unroll
            for (int jt = 0; jt < 8; jt++) sm += __expf(acc[jt][r] - mx);
            for (int o = 1; o < 16; o <<= 1) sm += __shfl_xor(sm, o, 32);
            float lse = mx + __logf(sm);
            int ml = wave * 16 + half * 8 + r;
            if ((lane & 15) == 0) slog[(size_t)bh * TOT + c * BS + ml] = lse;
#pragma unroll
            for (int jt = 0; jt < 8; jt++)
                s_probs[ml][jt * 16 + (lane & 15)] = (__bf16)__expf(acc[jt][r] - lse);
        }
    }
    __syncthreads();

    // ---- out = probs @ bv (64x64, K=128) ----
    if (wave < 4) {
        v8f acc2[4] = {zero, zero, zero, zero};
        int mrow = wave * 16 + (lane & 15);
        for (int kb = 0; kb < 128; kb += 32) {
            Frag16 a;
            a.v = *(const v16bf*)&s_probs[mrow][kb + half * 16];
#pragma unroll
            for (int et = 0; et < 4; et++) {
                Frag16 b;
                b.v = *(const v16bf*)&s_bvT[et * 16 + (lane & 15)][kb + half * 16];
                acc2[et] = wmma_bf16(a.v, b.v, acc2[et]);
            }
        }
#pragma unroll
        for (int et = 0; et < 4; et++)
#pragma unroll
            for (int r = 0; r < 8; r++) {
                int ml  = wave * 16 + half * 8 + r;
                int col = et * 16 + (lane & 15);
                so[((size_t)bh * TOT + c * BS + ml) * DHEAD + col] = acc2[et][r];
            }
    }
}

// ---------------- unsort + softmax-combine over hash rounds ----------------
__global__ __launch_bounds__(256) void combine_kernel(const float* __restrict__ so,
                                                      const float* __restrict__ slog,
                                                      const int* __restrict__ undo,
                                                      __bf16* __restrict__ out) {
    int i  = blockIdx.x * 256 + threadIdx.x;   // NHEAD*SLEN threads
    int s  = i & (SLEN - 1);
    int bh = i >> 12;
    const int* u = undo + (size_t)bh * TOT;
    float l[NHASH]; int idx[NHASH];
#pragma unroll
    for (int h = 0; h < NHASH; h++) {
        idx[h] = u[h * SLEN + s];
        l[h] = slog[(size_t)bh * TOT + idx[h]];
    }
    float mx = fmaxf(fmaxf(l[0], l[1]), fmaxf(l[2], l[3]));
    float w[NHASH], ws = 0.f;
#pragma unroll
    for (int h = 0; h < NHASH; h++) { w[h] = __expf(l[h] - mx); ws += w[h]; }
    float inv = 1.f / ws;
    for (int d = 0; d < DHEAD; d++) {
        float a = 0.f;
#pragma unroll
        for (int h = 0; h < NHASH; h++)
            a += so[((size_t)bh * TOT + idx[h]) * DHEAD + d] * w[h];
        out[(size_t)s * EMBD + bh * DHEAD + d] = (__bf16)(a * inv);
    }
}

// ---------------- x1 + x2 -> bf16 ----------------
__global__ __launch_bounds__(256) void addbf_kernel(const float* __restrict__ a,
                                                    const float* __restrict__ b,
                                                    __bf16* __restrict__ o) {
    int i = blockIdx.x * 256 + threadIdx.x;
    o[i] = (__bf16)(a[i] + b[i]);
}

// ---------------- host launch ----------------
extern "C" void kernel_launch(void* const* d_in, const int* in_sizes, int n_in,
                              void* d_out, int out_size, void* d_ws, size_t ws_size,
                              hipStream_t stream) {
    const int*   tokens  = (const int*)  d_in[0];
    const float* tok_emb = (const float*)d_in[1];
    const float* pos_emb = (const float*)d_in[2];
    const float* ln_f_g  = (const float*)d_in[3];
    const float* ln_f_b  = (const float*)d_in[4];
    const float* Wqk     = (const float*)d_in[5];
    const float* Wv      = (const float*)d_in[6];
    const float* Wo      = (const float*)d_in[7];
    const float* bo      = (const float*)d_in[8];
    const float* ln_g_g  = (const float*)d_in[9];
    const float* ln_g_b  = (const float*)d_in[10];
    const float* W1      = (const float*)d_in[11];
    const float* b1      = (const float*)d_in[12];
    const float* W2      = (const float*)d_in[13];
    const float* b2      = (const float*)d_in[14];
    const float* rot     = (const float*)d_in[15];
    const float* Wl      = (const float*)d_in[16];
    const float* bl      = (const float*)d_in[17];
    float* out = (float*)d_out;

    char* p = (char*)d_ws;
    auto carve = [&](size_t bytes) { char* r = p; p += (bytes + 255) & ~(size_t)255; return r; };
    __bf16* pWqk = (__bf16*)carve((size_t)4 * EMBD * EMBD * 2);
    __bf16* pWv  = (__bf16*)carve((size_t)4 * EMBD * EMBD * 2);
    __bf16* pWo  = (__bf16*)carve((size_t)4 * EMBD * EMBD * 2);
    __bf16* pW1  = (__bf16*)carve((size_t)4 * EMBD * FFND * 2);
    __bf16* pW2  = (__bf16*)carve((size_t)4 * FFND * EMBD * 2);
    __bf16* pWl  = (__bf16*)carve((size_t)EMBD * VOCAB * 2);
    float*  x1   = (float*) carve((size_t)SLEN * EMBD * 4);
    float*  x2   = (float*) carve((size_t)SLEN * EMBD * 4);
    float*  QKb  = (float*) carve((size_t)SLEN * EMBD * 4);
    float*  Vb   = (float*) carve((size_t)SLEN * EMBD * 4);
    __bf16* xbf  = (__bf16*)carve((size_t)SLEN * EMBD * 2);
    __bf16* hbf  = (__bf16*)carve((size_t)SLEN * FFND * 2);
    int*    bck  = (int*)   carve((size_t)NHEAD * TOT * 4);
    int*    srt  = (int*)   carve((size_t)NHEAD * TOT * 4);
    int*    und  = (int*)   carve((size_t)NHEAD * TOT * 4);
    float*  so   = (float*) carve((size_t)NHEAD * TOT * DHEAD * 4);
    float*  slog = (float*) carve((size_t)NHEAD * TOT * 4);

    // ---- pack all weights to WMMA B-fragment layout (bf16) ----
    for (int d = 0; d < 4; d++) {
        pack_kernel<<<1024, 256, 0, stream>>>(Wqk + (size_t)d * EMBD * EMBD, pWqk + (size_t)d * EMBD * EMBD, EMBD, EMBD);
        pack_kernel<<<1024, 256, 0, stream>>>(Wv  + (size_t)d * EMBD * EMBD, pWv  + (size_t)d * EMBD * EMBD, EMBD, EMBD);
        pack_kernel<<<1024, 256, 0, stream>>>(Wo  + (size_t)d * EMBD * EMBD, pWo  + (size_t)d * EMBD * EMBD, EMBD, EMBD);
        pack_kernel<<<4096, 256, 0, stream>>>(W1  + (size_t)d * EMBD * FFND, pW1  + (size_t)d * EMBD * FFND, EMBD, FFND);
        pack_kernel<<<4096, 256, 0, stream>>>(W2  + (size_t)d * FFND * EMBD, pW2  + (size_t)d * FFND * EMBD, FFND, EMBD);
    }
    pack_kernel<<<64000, 256, 0, stream>>>(Wl, pWl, EMBD, VOCAB);

    // ---- embedding (x1 = x2 = emb) ----
    embed_kernel<<<(SLEN * EMBD) / 256, 256, 0, stream>>>(tokens, tok_emb, pos_emb, x1, x2);

    for (int d = 0; d < 4; d++) {
        // attention branch: ln_f(x2)
        layernorm_bf16<<<SLEN, 256, 0, stream>>>(x2, ln_f_g + d * EMBD, ln_f_b + d * EMBD, xbf);
        wmma_gemm<<<dim3(EMBD / 128, SLEN / 64), 256, 0, stream>>>(
            xbf, pWqk + (size_t)d * EMBD * EMBD, nullptr, nullptr, QKb, nullptr, EMBD, EMBD, 0);
        wmma_gemm<<<dim3(EMBD / 128, SLEN / 64), 256, 0, stream>>>(
            xbf, pWv + (size_t)d * EMBD * EMBD, nullptr, nullptr, Vb, nullptr, EMBD, EMBD, 0);
        bucket_kernel<<<(NHEAD * NHASH * SLEN) / 256, 256, 0, stream>>>(
            QKb, rot + (size_t)d * DHEAD * NHASH * 32, bck);
        sort_kernel<<<NHEAD, 256, 0, stream>>>(bck, srt, und);
        attn_kernel<<<dim3(NCH, NHEAD), 256, 0, stream>>>(QKb, Vb, srt, so, slog);
        combine_kernel<<<(NHEAD * SLEN) / 256, 256, 0, stream>>>(so, slog, und, xbf);
        // y1 = x1 + attn @ Wo + bo   (written back into x1)
        wmma_gemm<<<dim3(EMBD / 128, SLEN / 64), 256, 0, stream>>>(
            xbf, pWo + (size_t)d * EMBD * EMBD, bo + d * EMBD, x1, x1, nullptr, EMBD, EMBD, 0);
        // FFN branch: ln_g(y1)
        layernorm_bf16<<<SLEN, 256, 0, stream>>>(x1, ln_g_g + d * EMBD, ln_g_b + d * EMBD, xbf);
        wmma_gemm<<<dim3(FFND / 128, SLEN / 64), 256, 0, stream>>>(
            xbf, pW1 + (size_t)d * EMBD * FFND, b1 + d * FFND, nullptr, nullptr, hbf, FFND, EMBD, 1);
        // y2 = x2 + gelu(h) @ W2 + b2  (written back into x2)
        wmma_gemm<<<dim3(EMBD / 128, SLEN / 64), 256, 0, stream>>>(
            hbf, pW2 + (size_t)d * FFND * EMBD, b2 + d * EMBD, x2, x2, nullptr, EMBD, FFND, 0);
    }

    // logits = (x1 + x2) @ Wl + bl
    addbf_kernel<<<(SLEN * EMBD) / 256, 256, 0, stream>>>(x1, x2, xbf);
    wmma_gemm<<<dim3(VOCAB / 128, SLEN / 64), 256, 0, stream>>>(
        xbf, pWl, bl, nullptr, out, nullptr, VOCAB, EMBD, 0);
}
